// MultichannelMultiheadAttention_51591147159623
// MI455X (gfx1250) — matrix operand order repaired and checked by reading, not compiled, not run on hardware
//
#include <hip/hip_runtime.h>
#include <hip/hip_bf16.h>

// Problem constants
#define NB    1024            // BINS (M=N=K of the big GEMMs, also seq len)
#define NNE   (1024*1024)     // elements per 1024x1024 matrix
#define BCC   8               // batch(2) * channels(4)
#define NHEAD 8
#define HD    128             // head dim

typedef __attribute__((ext_vector_type(16))) _Float16 v16h;
typedef __attribute__((ext_vector_type(8)))  _Float16 v8h;
typedef __attribute__((ext_vector_type(8)))  float    v8f;
typedef __attribute__((ext_vector_type(4)))  float    v4f;

// ---------------------------------------------------------------------------
// WMMA fragment loaders (wave32, V_WMMA_F32_16X16X32_F16 layouts per CDNA5 ISA)
//
// A (16x32, 16-bit): lane L holds row M = L&15; half-slot e holds
//   K = (e>>3)*16 + (L>>4)*8 + (e&7)  -> two contiguous 8-half (16B) chunks.
// B (32x16, 16-bit): lane L holds col N = L&15; half-slot e holds
//   K = (e>>3)*16 + (L>>4)*8 + (e&7)  -> same as A (two 8-half chunks).
// C/D (16x16 f32): lane L: N = L&15; vgpr r: M = (L>>4)*8 + r.
// ---------------------------------------------------------------------------

static __device__ __forceinline__ v16h load_a_f16(const _Float16* A, int ld,
                                                  int row, int k0, int hi) {
  const _Float16* p = A + (size_t)row * ld + k0 + hi * 8;
  v8h x0 = *(const v8h*)p;          // K = k0 + hi*8 .. +7
  v8h x1 = *(const v8h*)(p + 16);   // K = k0 + 16 + hi*8 .. +7
  v16h out;
#pragma unroll
  for (int i = 0; i < 8; ++i) { out[i] = x0[i]; out[8 + i] = x1[i]; }
  return out;
}

static __device__ __forceinline__ v16h load_a_f32(const float* A, int ld,
                                                  int row, int k0, int hi) {
  const float* p = A + (size_t)row * ld + k0 + hi * 8;
  v4f a0 = *(const v4f*)p;
  v4f a1 = *(const v4f*)(p + 4);
  v4f a2 = *(const v4f*)(p + 16);
  v4f a3 = *(const v4f*)(p + 20);
  v16h out;
#pragma unroll
  for (int i = 0; i < 4; ++i) {
    out[i]      = (_Float16)a0[i];
    out[4 + i]  = (_Float16)a1[i];
    out[8 + i]  = (_Float16)a2[i];
    out[12 + i] = (_Float16)a3[i];
  }
  return out;
}

static __device__ __forceinline__ v16h load_b_f16(const _Float16* B, int ld,
                                                  int row, int k0, int hi) {
  const _Float16* p = B + (size_t)row * ld + k0 + hi * 8;
  v8h x0 = *(const v8h*)p;
  v8h x1 = *(const v8h*)(p + 16);
  v16h out;
#pragma unroll
  for (int i = 0; i < 8; ++i) { out[i] = x0[i]; out[8 + i] = x1[i]; }
  return out;
}

static __device__ __forceinline__ v16h load_b_f32(const float* B, int ld,
                                                  int row, int k0, int hi) {
  const float* p = B + (size_t)row * ld + k0 + hi * 8;
  v4f a0 = *(const v4f*)p;
  v4f a1 = *(const v4f*)(p + 4);
  v4f a2 = *(const v4f*)(p + 16);
  v4f a3 = *(const v4f*)(p + 20);
  v16h out;
#pragma unroll
  for (int i = 0; i < 4; ++i) {
    out[i]      = (_Float16)a0[i];
    out[4 + i]  = (_Float16)a1[i];
    out[8 + i]  = (_Float16)a2[i];
    out[12 + i] = (_Float16)a3[i];
  }
  return out;
}

static __device__ __forceinline__ v8f wmma_f16(v16h a, v16h b, v8f c) {
  return __builtin_amdgcn_wmma_f32_16x16x32_f16(false, a, false, b,
                                                (short)0, c, false, false);
}

// ---------------------------------------------------------------------------
// Kernel 1/5: batched GEMM  C[z] = A[z] (1024x1024) * B[z&3]^T (f32 weights)
// A dtype templated (f32 for proj of x, f16 for attention output).
// Block = 8 waves, 128x128 tile; wave = 64(m) x 32(n); K stepped by 32.
// Next K-slab is prefetched (global_prefetch_b8) under the current WMMAs.
// ---------------------------------------------------------------------------
template <typename TA>
__global__ __launch_bounds__(256) void gemm_nt_kernel(const TA* __restrict__ A,
                                                      const float* __restrict__ Bw,
                                                      float* __restrict__ C) {
  const int z = blockIdx.z;
  const TA*    Az = A  + (size_t)z * NNE;
  const float* Bz = Bw + (size_t)(z & 3) * NNE;
  float*       Cz = C  + (size_t)z * NNE;

  const int wid  = threadIdx.x >> 5;
  const int lane = threadIdx.x & 31;
  const int lo   = lane & 15;
  const int hi   = lane >> 4;

  const int m0 = blockIdx.y * 128 + (wid >> 2) * 64;  // 4 m-frags
  const int n0 = blockIdx.x * 128 + (wid & 3) * 32;   // 2 n-frags

  v8f acc[4][2];
#pragma unroll
  for (int mf = 0; mf < 4; ++mf)
#pragma unroll
    for (int nf = 0; nf < 2; ++nf) acc[mf][nf] = (v8f)0.0f;

  for (int k0 = 0; k0 < NB; k0 += 32) {
    // Speculative prefetch of the next K slab (L2 -> WGP$) while WMMAs run.
    if (k0 + 32 < NB) {
      __builtin_prefetch(Az + (size_t)(m0 + lane) * NB + k0 + 32, 0, 1);
      __builtin_prefetch(Az + (size_t)(m0 + 32 + lane) * NB + k0 + 32, 0, 1);
      __builtin_prefetch(Bz + (size_t)(n0 + lane) * NB + k0 + 32, 0, 1);
    }

    v16h afr[4];
#pragma unroll
    for (int mf = 0; mf < 4; ++mf) {
      if constexpr (__is_same(TA, float))
        afr[mf] = load_a_f32((const float*)Az, NB, m0 + mf * 16 + lo, k0, hi);
      else
        afr[mf] = load_a_f16((const _Float16*)Az, NB, m0 + mf * 16 + lo, k0, hi);
    }
#pragma unroll
    for (int nf = 0; nf < 2; ++nf) {
      v16h bfr = load_b_f32(Bz, NB, n0 + nf * 16 + lo, k0, hi);
#pragma unroll
      for (int mf = 0; mf < 4; ++mf)
        acc[mf][nf] = wmma_f16(afr[mf], bfr, acc[mf][nf]);
    }
  }

#pragma unroll
  for (int mf = 0; mf < 4; ++mf)
#pragma unroll
    for (int nf = 0; nf < 2; ++nf)
#pragma unroll
      for (int r = 0; r < 8; ++r)
        Cz[(size_t)(m0 + mf * 16 + hi * 8 + r) * NB + n0 + nf * 16 + lo] =
            acc[mf][nf][r];
}

// ---------------------------------------------------------------------------
// Kernel 2/5: depthwise 1x3 conv along width + bias.
// P[bc][h][o] = b[c] + sum_k Y[bc][h][o-1+k] * wc[c][k]   (zero padded)
// Also emits f16 copy (used as V, layout [row=h][col=w]).
// ---------------------------------------------------------------------------
__global__ __launch_bounds__(256) void conv_bias_kernel(
    const float* __restrict__ Y, const float* __restrict__ wc,
    const float* __restrict__ bias, float* __restrict__ P,
    _Float16* __restrict__ Ph) {
  const size_t idx = (size_t)blockIdx.x * blockDim.x + threadIdx.x;
  const int o  = (int)(idx & 1023);
  const int bc = (int)(idx >> 20);
  const int c  = bc & 3;

  const float w0 = wc[c * 3 + 0], w1 = wc[c * 3 + 1], w2 = wc[c * 3 + 2];
  const float ym = (o > 0)    ? Y[idx - 1] : 0.0f;
  const float y0 = Y[idx];
  const float yp = (o < 1023) ? Y[idx + 1] : 0.0f;
  const float v = bias[c] + ym * w0 + y0 * w1 + yp * w2;
  P[idx]  = v;
  Ph[idx] = (_Float16)v;
}

// ---------------------------------------------------------------------------
// Kernel 3/5: head split + RoPE -> Qh[u][w][d] f16, u = bc*8 + head.
// ph[u][w][d] = P[bc][head*128 + d][w]; rotate interleaved pairs of first
// 64 dims with angle w * 10000^(-t/32); pass-through dims 64..127.
// ---------------------------------------------------------------------------
__global__ __launch_bounds__(256) void rope_kernel(const float* __restrict__ P,
                                                   _Float16* __restrict__ Qh) {
  const size_t tid = (size_t)blockIdx.x * blockDim.x + threadIdx.x;
  const int t = (int)(tid & 63);          // pair index, d = 2t, 2t+1
  const int w = (int)((tid >> 6) & 1023); // sequence position
  const int u = (int)(tid >> 16);         // 0..63 = bc*8 + head
  const int head = u & 7;
  const int bc   = u >> 3;

  const float* src = P + (size_t)bc * NNE + (size_t)(head * HD + 2 * t) * NB + w;
  const float p0 = src[0];
  const float p1 = src[NB];

  _Float16* dst = Qh + (size_t)u * (NB * HD) + (size_t)w * HD + 2 * t;
  if (t < 32) {
    const float ang = (float)w * __powf(10000.0f, -(float)t * (1.0f / 32.0f));
    float sn, cs;
    __sincosf(ang, &sn, &cs);
    dst[0] = (_Float16)(p0 * cs - p1 * sn);
    dst[1] = (_Float16)(p1 * cs + p0 * sn);
  } else {
    dst[0] = (_Float16)p0;
    dst[1] = (_Float16)p1;
  }
}

// ---------------------------------------------------------------------------
// Kernel 4/5: flash attention per (bc, head). q == k == Qh, v == Ph.
// Grid (64 units, 8 q-tiles); block = 8 waves; wave owns 16 query rows.
// S tile = 16q x 32key (2 WMMA n-frags, K over d in 4 steps of 32).
// Online softmax; probs re-shaped D->A layout through per-wave LDS tile;
// O (16q x 128d) accumulated with 8 WMMAs per key step.
// Output stored transposed: Ah[bc][head*128 + d][q] (f16).
// ---------------------------------------------------------------------------
__global__ __launch_bounds__(256) void attention_kernel(
    const _Float16* __restrict__ Qh, const _Float16* __restrict__ Vh,
    _Float16* __restrict__ Ah) {
  __shared__ __align__(16) _Float16 slds[8][16][32];  // per-wave 16x32 prob tile

  const int u    = blockIdx.x;          // bc*8 + head
  const int wid  = threadIdx.x >> 5;
  const int lane = threadIdx.x & 31;
  const int lo   = lane & 15;
  const int hi   = lane >> 4;
  const int q0   = blockIdx.y * 128 + wid * 16;

  const int bc = u >> 3, head = u & 7;
  const _Float16* Qb = Qh + (size_t)u * (NB * HD);                  // [w][d]
  const _Float16* Vb = Vh + (size_t)bc * NNE + (size_t)(head * HD) * NB; // [d][w]
  _Float16*       Ab = Ah + (size_t)bc * NNE + (size_t)(head * HD) * NB; // [d][q]

  // Preload 4 Q A-fragments (16 q rows x full d=128)
  v16h qa[4];
#pragma unroll
  for (int kf = 0; kf < 4; ++kf)
    qa[kf] = load_a_f16(Qb, HD, q0 + lo, kf * 32, hi);

  v8f O[8];
#pragma unroll
  for (int df = 0; df < 8; ++df) O[df] = (v8f)0.0f;
  float m8[8], l8[8];
#pragma unroll
  for (int r = 0; r < 8; ++r) { m8[r] = -1.0e30f; l8[r] = 0.0f; }

  const float inv_scale = 0.03125f;  // 1/sqrt(1024)

  for (int j0 = 0; j0 < NB; j0 += 32) {
    // Prefetch next key tile (K rows) and next V column slab.
    if (j0 + 32 < NB) {
      __builtin_prefetch(Qb + (size_t)(j0 + 32 + lane) * HD, 0, 1);
      __builtin_prefetch(Vb + (size_t)(lane * 4) * NB + j0 + 32, 0, 1);
    }

    // ---- S = Q * K^T for 32 keys (K == Q buffer) ----
    v8f s0 = (v8f)0.0f, s1 = (v8f)0.0f;
#pragma unroll
    for (int kf = 0; kf < 4; ++kf) {
      v16h b0 = load_b_f16(Qb, HD, j0 + lo,      kf * 32, hi);
      v16h b1 = load_b_f16(Qb, HD, j0 + 16 + lo, kf * 32, hi);
      s0 = wmma_f16(qa[kf], b0, s0);
      s1 = wmma_f16(qa[kf], b1, s1);
    }

    // ---- online softmax (rows M = hi*8 + r; cols across lo lanes) ----
    float p0[8], p1[8], tmax[8];
#pragma unroll
    for (int r = 0; r < 8; ++r) {
      p0[r] = s0[r] * inv_scale;
      p1[r] = s1[r] * inv_scale;
      tmax[r] = fmaxf(p0[r], p1[r]);
    }
#pragma unroll
    for (int mask = 1; mask <= 8; mask <<= 1)
#pragma unroll
      for (int r = 0; r < 8; ++r)
        tmax[r] = fmaxf(tmax[r], __shfl_xor(tmax[r], mask, 32));

    float alpha[8];
#pragma unroll
    for (int r = 0; r < 8; ++r) {
      const float mn = fmaxf(m8[r], tmax[r]);
      alpha[r] = __expf(m8[r] - mn);
      m8[r] = mn;
      p0[r] = __expf(p0[r] - mn);
      p1[r] = __expf(p1[r] - mn);
    }
    float tsum[8];
#pragma unroll
    for (int r = 0; r < 8; ++r) tsum[r] = p0[r] + p1[r];
#pragma unroll
    for (int mask = 1; mask <= 8; mask <<= 1)
#pragma unroll
      for (int r = 0; r < 8; ++r) tsum[r] += __shfl_xor(tsum[r], mask, 32);
#pragma unroll
    for (int r = 0; r < 8; ++r) l8[r] = l8[r] * alpha[r] + tsum[r];
#pragma unroll
    for (int df = 0; df < 8; ++df)
#pragma unroll
      for (int r = 0; r < 8; ++r) O[df][r] *= alpha[r];

    // ---- D-layout probs -> A-layout via per-wave LDS tile ----
#pragma unroll
    for (int r = 0; r < 8; ++r) {
      slds[wid][hi * 8 + r][lo]      = (_Float16)p0[r];
      slds[wid][hi * 8 + r][16 + lo] = (_Float16)p1[r];
    }
    // LDS is serviced in-order per wave on CDNA5; wait for stores, block
    // compiler reordering of the subsequent reads.
    asm volatile("s_wait_dscnt 0" ::: "memory");
    v16h pa;
    {
      v8h x0 = *(const v8h*)&slds[wid][lo][hi * 8];
      v8h x1 = *(const v8h*)&slds[wid][lo][16 + hi * 8];
#pragma unroll
      for (int i = 0; i < 8; ++i) { pa[i] = x0[i]; pa[8 + i] = x1[i]; }
    }
    asm volatile("s_wait_dscnt 0" ::: "memory");

    // ---- O += P * V ----
#pragma unroll
    for (int df = 0; df < 8; ++df) {
      v16h vb = load_b_f16(Vb, NB, df * 16 + lo, j0, hi);
      O[df] = wmma_f16(pa, vb, O[df]);
    }
  }

  // ---- normalize and store transposed: Ah[d][q] ----
  float invl[8];
#pragma unroll
  for (int r = 0; r < 8; ++r) invl[r] = 1.0f / l8[r];
#pragma unroll
  for (int df = 0; df < 8; ++df) {
    v8h hv;
#pragma unroll
    for (int r = 0; r < 8; ++r) hv[r] = (_Float16)(O[df][r] * invl[r]);
    *(v8h*)(Ab + (size_t)(df * 16 + lo) * NB + q0 + hi * 8) = hv;
  }
}

// ---------------------------------------------------------------------------
// Host-side launch
// ---------------------------------------------------------------------------
extern "C" void kernel_launch(void* const* d_in, const int* in_sizes, int n_in,
                              void* d_out, int out_size, void* d_ws,
                              size_t ws_size, hipStream_t stream) {
  const float* x      = (const float*)d_in[0];  // (2,4,1024,1024)
  const float* w_qlin = (const float*)d_in[1];  // (4,1024,1024)
  const float* w_qcnv = (const float*)d_in[2];  // (4,1,1,3)
  const float* b_qcnv = (const float*)d_in[3];  // (4,)
  const float* w_out  = (const float*)d_in[4];  // (4,1024,1024)
  float* out = (float*)d_out;                   // (2,4,1024,1024)

  const size_t BCNN = (size_t)BCC * NNE;
  float*    Y   = (float*)d_ws;                 // pre-conv GEMM out (f32)
  float*    P   = Y + BCNN;                     // post-conv proj    (f32)
  _Float16* Ph  = (_Float16*)(P + BCNN);        // V, f16 [bc][h][w]
  _Float16* Qh  = Ph + BCNN;                    // rope(Q)=rope(K) [u][w][d]
  _Float16* Ah  = Qh + BCNN;                    // attn out f16 [bc][h][w]
  // total ws use: 2*4B*8M + 3*2B*8M = 112 MiB

  (void)in_sizes; (void)n_in; (void)out_size; (void)ws_size;

  // 1) proj GEMM: Y[bc] = x[bc] * w_qlin[c]^T
  gemm_nt_kernel<float><<<dim3(8, 8, 8), 256, 0, stream>>>(x, w_qlin, Y);
  // 2) depthwise conv + bias -> P (f32) and Ph (f16 V)
  conv_bias_kernel<<<(unsigned)(BCNN / 256), 256, 0, stream>>>(Y, w_qcnv,
                                                               b_qcnv, P, Ph);
  // 3) head split + RoPE -> Qh
  rope_kernel<<<16384, 256, 0, stream>>>(P, Qh);
  // 4) flash attention -> Ah (transposed f16)
  attention_kernel<<<dim3(64, 8), 256, 0, stream>>>(Qh, Ph, Ah);
  // 5) output GEMM: out[bc] = Ah[bc] * w_out[c]^T
  gemm_nt_kernel<_Float16><<<dim3(8, 8, 8), 256, 0, stream>>>(Ah, w_out, out);
}
